// PTSA_45148696216169
// MI455X (gfx1250) — matrix operand order repaired and checked by reading, not compiled, not run on hardware
//
#include <hip/hip_runtime.h>
#include <hip/hip_bf16.h>
#include <stdint.h>

// ---------------- problem constants (from reference) ----------------
#define B_   4
#define L_   4096
#define C_   384      // K and N of every GEMM
#define H_   6
#define D_   64
#define LP_  2048     // parent scale length
#define LC_  8192     // child scale length
#define NC_  18       // candidates per query: 13 band + 3 parent + 2 child
#define WN_  (C_*C_)  // 147456 elements per weight matrix

// ---------------- WMMA types ----------------
typedef __attribute__((ext_vector_type(16))) __bf16 v16bf;
typedef __attribute__((ext_vector_type(8)))  float  v8f;

union Frag16 { uint4 u[2]; v16bf v; };

__device__ __forceinline__ unsigned short f2bf(float f) {
    union { float f; unsigned int u; } x; x.f = f;
    unsigned int u = x.u;
    unsigned int r = u + 0x7fffu + ((u >> 16) & 1u);   // round-to-nearest-even
    return (unsigned short)(r >> 16);
}
__device__ __forceinline__ float bf2f(unsigned int lo16) {
    union { unsigned int u; float f; } x; x.u = lo16 << 16;
    return x.f;
}

// CDNA5 async global->LDS copy (ASYNCcnt-tracked); vdst = LDS byte address VGPR
__device__ __forceinline__ void async_g2l_b128(unsigned lds_byte, const void* gaddr) {
    asm volatile("global_load_async_to_lds_b128 %0, %1, off"
                 :: "v"(lds_byte), "v"((unsigned long long)(uintptr_t)gaddr)
                 : "memory");
}
__device__ __forceinline__ void wait_async0() {
    asm volatile("s_wait_asynccnt 0x0" ::: "memory");
}

// ---------------- weight f32 -> bf16 conversion ----------------
__global__ __launch_bounds__(256) void cvt_w_kernel(
    const float* __restrict__ Wq, const float* __restrict__ Wk,
    const float* __restrict__ Wv, const float* __restrict__ Wp,
    unsigned short* __restrict__ out)
{
    int i = blockIdx.x * 256 + threadIdx.x;        // 4*WN_ total
    const float* src = (i < WN_) ? Wq : (i < 2*WN_) ? Wk : (i < 3*WN_) ? Wv : Wp;
    out[i] = f2bf(src[i % WN_]);
}

// ---------------- layernorm: one wave per row ----------------
__global__ __launch_bounds__(256) void ln_kernel(
    const float* __restrict__ X, const float* __restrict__ gamma,
    const float* __restrict__ beta, float* __restrict__ Y)
{
    const int wid  = threadIdx.x >> 5;
    const int lane = threadIdx.x & 31;
    const size_t row = (size_t)blockIdx.x * 8 + wid;   // B_*L_ rows total
    const float* xr = X + row * C_;
    float v[12]; float sum = 0.f, sq = 0.f;
#pragma unroll
    for (int i = 0; i < 12; ++i) { v[i] = xr[lane + 32*i]; sum += v[i]; sq += v[i]*v[i]; }
#pragma unroll
    for (int o = 16; o > 0; o >>= 1) { sum += __shfl_xor(sum, o, 32); sq += __shfl_xor(sq, o, 32); }
    const float mean = sum * (1.f / C_);
    const float var  = sq  * (1.f / C_) - mean * mean;
    const float rs   = rsqrtf(var + 1e-5f);
    float* yr = Y + row * C_;
#pragma unroll
    for (int i = 0; i < 12; ++i) {
        int col = lane + 32*i;
        yr[col] = (v[i] - mean) * rs * gamma[col] + beta[col];
    }
}

// ---------------- WMMA GEMM: Y[M,384] = X_f32[M,384] @ W_bf16^T ----------------
// Block = 256 threads (8 waves). Tile: 64 M-rows x 96 N-cols (blockIdx.y of 4).
// A tile f32->bf16 staged by VALU; W sub-tile staged via async global->LDS.
// wave (wid&3) -> 16-row M sub-tile, (wid>>2) -> 3 of the 6 N-tiles.
#define APITCH 392   // bf16 elems per LDS A row (16B-aligned rows: 392*2=784=16*49)

__global__ __launch_bounds__(256) void gemm_rt_bf16(
    const float* __restrict__ X, const unsigned short* __restrict__ W,
    void* __restrict__ Yraw, int outIsBf16)
{
    __shared__ __align__(16) unsigned short At[64 * APITCH];     //  50,176 B
    __shared__ __align__(16) unsigned short Wl[96 * C_];         //  73,728 B
    const int tid   = threadIdx.x;
    const int mBase = blockIdx.x * 64;
    const int nBase = blockIdx.y * 96;

    // ---- stage W sub-tile (96 x 384 bf16 = 4608 x 16B) via async copy ----
    {
        const unsigned short* wsrc = W + (size_t)nBase * C_;
        const unsigned ldsbase = (unsigned)(uintptr_t)(&Wl[0]);
#pragma unroll
        for (int j = 0; j < 18; ++j) {
            const int idx = tid + j * 256;
            async_g2l_b128(ldsbase + idx * 16, wsrc + idx * 8);
        }
    }

    // ---- stage A: 64 rows x 384 f32 -> bf16 (6144 float4, 24 per thread) ----
#pragma unroll
    for (int i = 0; i < 24; ++i) {
        int f4 = tid + i * 256;
        int r  = f4 / 96;
        int c4 = (f4 % 96) * 4;
        const float4 xv = ((const float4*)(X + (size_t)(mBase + r) * C_))[c4 >> 2];
        unsigned short* dst = &At[r * APITCH + c4];
        dst[0] = f2bf(xv.x); dst[1] = f2bf(xv.y); dst[2] = f2bf(xv.z); dst[3] = f2bf(xv.w);
    }
    wait_async0();
    __syncthreads();

    const int wid  = tid >> 5;
    const int lane = tid & 31;
    const int msub   = wid & 3;      // 0..3  -> rows 16*msub within the block tile
    const int ngroup = wid >> 2;     // 0..1  -> local N-tiles [3*ngroup, 3*ngroup+3)
    const int mrow   = lane & 15;    // row within 16x16 tile (A) / N within tile (B)
    const int grp    = lane >> 4;    // lane group

    // A fragments: ISA 16-bit A layout -> group g holds K {g*8..g*8+7, 16+g*8..16+g*8+7}
    Frag16 af[12];
    const unsigned short* arow = &At[(msub * 16 + mrow) * APITCH];
#pragma unroll
    for (int kc = 0; kc < 12; ++kc) {
        const unsigned short* p0 = arow + kc * 32 + grp * 8;
        af[kc].u[0] = *(const uint4*)p0;          // K = kc*32 + grp*8 .. +7
        af[kc].u[1] = *(const uint4*)(p0 + 16);   // K = kc*32 + 16 + grp*8 .. +7
    }

#pragma unroll
    for (int ntl = 0; ntl < 3; ++ntl) {
        const int ntLoc = ngroup * 3 + ntl;           // 0..5 within block
        const int nt    = (nBase >> 4) + ntLoc;       // global N-tile
        // B fragment source: lane group g holds K {g*8..g*8+7, 16+g*8..16+g*8+7} of col n=lane&15
        const unsigned short* wrow = &Wl[(ntLoc * 16 + mrow) * C_];

        v8f acc = {};
        Frag16 bfr[2];
        bfr[0].u[0] = *(const uint4*)(wrow + grp * 8);
        bfr[0].u[1] = *(const uint4*)(wrow + grp * 8 + 16);
#pragma unroll
        for (int kc = 0; kc < 12; ++kc) {
            if (kc + 1 < 12) {                         // prefetch next B fragment
                const unsigned short* pn = wrow + (kc + 1) * 32 + grp * 8;
                bfr[(kc + 1) & 1].u[0] = *(const uint4*)pn;
                bfr[(kc + 1) & 1].u[1] = *(const uint4*)(pn + 16);
            }
            acc = __builtin_amdgcn_wmma_f32_16x16x32_bf16(
                false, af[kc].v, false, bfr[kc & 1].v, (short)0, acc, false, false);
        }
        // D layout: lane = (m>=8?16:0)+n ; VGPR r -> m = r + 8*(lane>>4)
        const int n  = lane & 15;
        const int mo = (lane >> 4) * 8;
        const size_t rbase = (size_t)(mBase + msub * 16 + mo);
        if (outIsBf16) {
            unsigned short* Y = (unsigned short*)Yraw;
#pragma unroll
            for (int r = 0; r < 8; ++r)
                Y[(rbase + r) * C_ + nt * 16 + n] = f2bf(acc[r]);
        } else {
            float* Y = (float*)Yraw;
#pragma unroll
            for (int r = 0; r < 8; ++r)
                Y[(rbase + r) * C_ + nt * 16 + n] = acc[r];
        }
    }
}

// ---------------- sparse attention: one wave per (b,h,t) ----------------
// Kb/Vb are concatenated [scale1 (B*L rows) | scale2 (B*LP rows) | scale0 (B*LC rows)]
__global__ __launch_bounds__(256) void attn_kernel(
    const unsigned short* __restrict__ Qb, const unsigned short* __restrict__ Kb,
    const unsigned short* __restrict__ Vb, float* __restrict__ Oattn)
{
    const int wid  = threadIdx.x >> 5;
    const int lane = threadIdx.x & 31;
    const int qid  = blockIdx.x * 8 + wid;        // b*(H*L) + h*L + t
    const int b    = qid / (H_ * L_);
    const int rem  = qid - b * (H_ * L_);
    const int h    = rem / L_;
    const int t    = rem - h * L_;

    // each lane owns dims {2*lane, 2*lane+1} of this head
    const unsigned int qu =
        ((const unsigned int*)(Qb + ((size_t)(b * L_ + t)) * C_ + h * D_))[lane];
    const float q0 = bf2f(qu & 0xffffu), q1 = bf2f(qu >> 16);

    int rows[NC_];
    {
        const int base1 = b * L_;
#pragma unroll
        for (int c = 0; c < 13; ++c) {
            int j = t - 6 + c;
            j = j < 0 ? 0 : (j > L_ - 1 ? L_ - 1 : j);
            rows[c] = base1 + j;
        }
        int p = t >> 1; if (p > LP_ - 1) p = LP_ - 1;
        const int base2 = B_ * L_ + b * LP_;
        rows[13] = base2 + p;
        rows[14] = base2 + (p > 0 ? p - 1 : 0);
        rows[15] = base2 + (p < LP_ - 1 ? p + 1 : LP_ - 1);
        const int base0 = B_ * L_ + B_ * LP_ + b * LC_;
        int j0 = 2 * t;     if (j0 > LC_ - 1) j0 = LC_ - 1;
        int j1 = 2 * t + 1; if (j1 > LC_ - 1) j1 = LC_ - 1;
        rows[16] = base0 + j0;
        rows[17] = base0 + j1;
    }

    float s[NC_];
#pragma unroll
    for (int c = 0; c < NC_; ++c) {
        const unsigned int ku =
            ((const unsigned int*)(Kb + (size_t)rows[c] * C_ + h * D_))[lane];
        float part = q0 * bf2f(ku & 0xffffu) + q1 * bf2f(ku >> 16);
#pragma unroll
        for (int o = 16; o > 0; o >>= 1) part += __shfl_xor(part, o, 32);
        s[c] = part * 0.125f;   // 1/sqrt(64)
    }

    // top-16 of 18 == drop the two smallest
    int i1 = 0; float m1 = s[0];
#pragma unroll
    for (int c = 1; c < NC_; ++c) if (s[c] < m1) { m1 = s[c]; i1 = c; }
    int i2 = -1; float m2 = 3.4e38f;
#pragma unroll
    for (int c = 0; c < NC_; ++c) if (c != i1 && s[c] < m2) { m2 = s[c]; i2 = c; }

    float mx = -3.4e38f;
#pragma unroll
    for (int c = 0; c < NC_; ++c) if (c != i1 && c != i2 && s[c] > mx) mx = s[c];
    float w[NC_]; float sum = 0.f;
#pragma unroll
    for (int c = 0; c < NC_; ++c) {
        float e = (c == i1 || c == i2) ? 0.f : __expf(s[c] - mx);
        w[c] = e; sum += e;
    }
    const float inv = 1.f / sum;

    float o0 = 0.f, o1 = 0.f;
#pragma unroll
    for (int c = 0; c < NC_; ++c) {
        if (w[c] == 0.f) continue;
        const unsigned int vu =
            ((const unsigned int*)(Vb + (size_t)rows[c] * C_ + h * D_))[lane];
        const float ww = w[c] * inv;
        o0 += ww * bf2f(vu & 0xffffu);
        o1 += ww * bf2f(vu >> 16);
    }
    float2* op = (float2*)(Oattn + ((size_t)(b * L_ + t)) * C_ + h * D_);
    op[lane] = make_float2(o0, o1);
}

// ---------------- host-side orchestration ----------------
extern "C" void kernel_launch(void* const* d_in, const int* in_sizes, int n_in,
                              void* d_out, int out_size, void* d_ws, size_t ws_size,
                              hipStream_t stream) {
    const float* x     = (const float*)d_in[0];
    const float* pk0   = (const float*)d_in[1];
    const float* pk1   = (const float*)d_in[2];
    const float* pk2   = (const float*)d_in[3];
    const float* pv0   = (const float*)d_in[4];
    const float* pv1   = (const float*)d_in[5];
    const float* pv2   = (const float*)d_in[6];
    const float* Wq    = (const float*)d_in[7];
    const float* Wk    = (const float*)d_in[8];
    const float* Wv    = (const float*)d_in[9];
    const float* Wp    = (const float*)d_in[10];
    const float* gamma = (const float*)d_in[11];
    const float* beta  = (const float*)d_in[12];

    // workspace layout (all offsets 256B-aligned)
    char* ws = (char*)d_ws;
    const size_t OFF_WB   = 0;                                   // 4*WN_ bf16
    const size_t OFF_XLN  = OFF_WB  + (size_t)4 * WN_ * 2;        // B*L*C f32
    const size_t OFF_QB   = OFF_XLN + (size_t)B_ * L_ * C_ * 4;   // B*L*C bf16
    const size_t OFF_KB   = OFF_QB  + (size_t)B_ * L_ * C_ * 2;   // (B*L+B*LP+B*LC)*C bf16
    const size_t KV_ROWS  = (size_t)B_ * (L_ + LP_ + LC_);        // 57344
    const size_t OFF_VB   = OFF_KB  + KV_ROWS * C_ * 2;
    const size_t OFF_ATT  = OFF_VB  + KV_ROWS * C_ * 2;           // B*L*C f32

    unsigned short* wb  = (unsigned short*)(ws + OFF_WB);
    float*          xln = (float*)         (ws + OFF_XLN);
    unsigned short* qb  = (unsigned short*)(ws + OFF_QB);
    unsigned short* kb  = (unsigned short*)(ws + OFF_KB);
    unsigned short* vb  = (unsigned short*)(ws + OFF_VB);
    float*          att = (float*)         (ws + OFF_ATT);

    const unsigned short* wq = wb;
    const unsigned short* wk = wb + (size_t)1 * WN_;
    const unsigned short* wv = wb + (size_t)2 * WN_;
    const unsigned short* wp = wb + (size_t)3 * WN_;

    // 1) weights -> bf16
    cvt_w_kernel<<<(4 * WN_) / 256, 256, 0, stream>>>(Wq, Wk, Wv, Wp, wb);
    // 2) layernorm
    ln_kernel<<<(B_ * L_) / 8, 256, 0, stream>>>(x, gamma, beta, xln);
    // 3) projections (WMMA GEMMs), Q/K/V stored bf16
    gemm_rt_bf16<<<dim3((B_ * L_) / 64, 4), 256, 0, stream>>>(xln, wq, qb, 1);
    gemm_rt_bf16<<<dim3((B_ * L_) / 64, 4), 256, 0, stream>>>(pk1, wk, kb, 1);
    gemm_rt_bf16<<<dim3((B_ * LP_) / 64, 4), 256, 0, stream>>>(pk2, wk, kb + (size_t)B_*L_*C_, 1);
    gemm_rt_bf16<<<dim3((B_ * LC_) / 64, 4), 256, 0, stream>>>(pk0, wk, kb + (size_t)B_*(L_+LP_)*C_, 1);
    gemm_rt_bf16<<<dim3((B_ * L_) / 64, 4), 256, 0, stream>>>(pv1, wv, vb, 1);
    gemm_rt_bf16<<<dim3((B_ * LP_) / 64, 4), 256, 0, stream>>>(pv2, wv, vb + (size_t)B_*L_*C_, 1);
    gemm_rt_bf16<<<dim3((B_ * LC_) / 64, 4), 256, 0, stream>>>(pv0, wv, vb + (size_t)B_*(L_+LP_)*C_, 1);
    // 4) sparse attention (top-16 of 18)
    attn_kernel<<<(B_ * H_ * L_) / 8, 256, 0, stream>>>(qb, kb, vb, att);
    // 5) output projection -> f32 d_out
    gemm_rt_bf16<<<dim3((B_ * L_) / 64, 4), 256, 0, stream>>>(att, wp, (float*)d_out, 0);
}
